// SelfAttention_qkv_8134668059253
// MI455X (gfx1250) — matrix operand order repaired and checked by reading, not compiled, not run on hardware
//
#include <hip/hip_runtime.h>

// ---------------------------------------------------------------------------
// Causal MHA forward for MI455X (gfx1250, wave32, WMMA bf16 -> f32 accum)
// Pipeline:
//   0) convert x -> bf16; convert+transpose Wq/Wk/Wv/Wo -> bf16 [n][k]
//   1) GEMM (double-buffered async-to-LDS staging) -> Q,K [h][n][64], V^T
//   2) flash attention (online softmax, causal), WMMA for QK^T and PV
//   3) GEMM out = attn @ Wo + bo (f32 out)
// ---------------------------------------------------------------------------

#define SEQ   4096
#define DIMM  1024
#define HEADS 16
#define DHEAD 64

#define CDNA5_TRY_ASYNC 1   // global_load_async_to_lds_b128 staging

typedef __attribute__((ext_vector_type(16))) __bf16 v16bf;
typedef __attribute__((ext_vector_type(8)))  float  v8f;
typedef __attribute__((ext_vector_type(2)))  __bf16 bf16x2;
typedef __attribute__((ext_vector_type(4)))  unsigned uint4v;

static __device__ __forceinline__ __bf16 f2bf(float f) {
  unsigned u = __builtin_bit_cast(unsigned, f);
  u += 0x7FFFu + ((u >> 16) & 1u);               // round-to-nearest-even
  unsigned short s = (unsigned short)(u >> 16);
  return __builtin_bit_cast(__bf16, s);
}

// packed f32x2 -> bf16x2 (v_cvt_pk_bf16_f32 when the target has it)
static __device__ __forceinline__ bf16x2 f2bf2(float a, float b) {
#if __has_builtin(__builtin_amdgcn_cvt_pk_bf16_f32)
  auto t = __builtin_amdgcn_cvt_pk_bf16_f32(a, b);
  return __builtin_bit_cast(bf16x2, t);
#else
  bf16x2 r; r[0] = f2bf(a); r[1] = f2bf(b); return r;
#endif
}

// A-fragment K index pattern (16-bit A 16x32): VGPR v holds K pair
//   k = (v<4 ? 0 : 16) + (v&3)*2 + hi*8
static __device__ __forceinline__ int a_kpat(int v, int hi) {
  return ((v >> 2) << 4) | ((v & 3) << 1) | (hi << 3);
}

// 16-byte global -> LDS copy. ASYNCcnt-tracked DMA, bypasses VGPRs
// (CDNA5 ISA). LDS operand = as3 offset (low 32 bits).
static __device__ __forceinline__ void ldsCopy16(__bf16* l, const __bf16* g) {
#if CDNA5_TRY_ASYNC
  unsigned loff = (unsigned)(size_t)l;
  asm volatile("global_load_async_to_lds_b128 %0, %1, off"
               :: "v"(loff), "v"(g) : "memory");
#else
  *(uint4v*)l = *(const uint4v*)g;
#endif
}
static __device__ __forceinline__ void ldsCopyWait() {
#if CDNA5_TRY_ASYNC
  asm volatile("s_wait_asynccnt 0" ::: "memory");
#endif
}

// ---------------------------------------------------------------------------
// Kernel 0a: elementwise f32 -> bf16 (2 elems/thread, packed convert)
// ---------------------------------------------------------------------------
__global__ __launch_bounds__(256) void cvt_bf16_kernel(
    const float* __restrict__ src, __bf16* __restrict__ dst, int n) {
  int i = (blockIdx.x * 256 + threadIdx.x) * 2;
  if (i + 1 < n)
    *(bf16x2*)&dst[i] = f2bf2(src[i], src[i + 1]);
}

// ---------------------------------------------------------------------------
// Kernel 0b: f32 [K][N] -> bf16 transposed [N][K]  (32x32 LDS tile)
// ---------------------------------------------------------------------------
__global__ __launch_bounds__(256) void cvt_transpose_kernel(
    const float* __restrict__ src, __bf16* __restrict__ dst) {
  __shared__ __bf16 t[32][33];
  const int bx = blockIdx.x * 32, by = blockIdx.y * 32;
  const int tx = threadIdx.x & 31, ty = threadIdx.x >> 5;   // 32 x 8
  #pragma unroll
  for (int i = 0; i < 32; i += 8)
    t[ty + i][tx] = f2bf(src[(size_t)(by + ty + i) * DIMM + bx + tx]);
  __syncthreads();
  #pragma unroll
  for (int i = 0; i < 32; i += 8)
    dst[(size_t)(bx + ty + i) * DIMM + by + tx] = t[tx][ty + i];
}

// ---------------------------------------------------------------------------
// stage one 64x32 A tile + 64x32 B tile into LDS (async b128 copies)
// ---------------------------------------------------------------------------
static __device__ __forceinline__ void stage_tiles(
    __bf16* AsB, __bf16* BsB, const __bf16* __restrict__ A,
    const __bf16* __restrict__ Bt, int m0, int n0, int kt, int tid) {
  #pragma unroll
  for (int i = 0; i < 2; ++i) {
    int c = tid + i * 128;                // 256 chunks per tile
    int row = c >> 2, part = (c & 3) * 8;
    ldsCopy16(&AsB[row * 40 + part], &A[(size_t)(m0 + row) * DIMM + kt + part]);
    ldsCopy16(&BsB[row * 40 + part], &Bt[(size_t)(n0 + row) * DIMM + kt + part]);
  }
}

// ---------------------------------------------------------------------------
// Kernel 1: projection GEMM  C[4096,1024] = A(bf16) @ Bt(bf16 [n][k])^T
//   vmode 0: dst[h][n][64] = C*scale        (Q, K)
//   vmode 1: dst[h][64][n] = C (transposed) (V)
// Block: 128 thr (4 waves), tile 64x64, wave 32x32 (2x2 WMMA), k-step 32.
// Double-buffered: async DMA of tile k+1 overlaps WMMA on tile k.
// ---------------------------------------------------------------------------
__global__ __launch_bounds__(128) void qkv_gemm_kernel(
    const __bf16* __restrict__ A, const __bf16* __restrict__ Bt,
    __bf16* __restrict__ dst, float scale, int vmode)
{
  __shared__ __bf16 As[2][64 * 40];   // [row][k], stride 40 (16B-aligned rows)
  __shared__ __bf16 Bs[2][64 * 40];   // [col][k]

  const int tid  = threadIdx.x;
  const int lane = tid & 31;
  const int w    = tid >> 5;
  const int nl   = lane & 15;
  const int hi   = lane >> 4;
  const int m0   = blockIdx.y * 64;
  const int n0   = blockIdx.x * 64;
  const int wm   = (w >> 1) * 32;
  const int wn   = (w & 1) * 32;

  v8f acc[2][2] = {};

  stage_tiles(As[0], Bs[0], A, Bt, m0, n0, 0, tid);
  ldsCopyWait();
  __syncthreads();

  for (int kt = 0; kt < DIMM; kt += 32) {
    const int cur = (kt >> 5) & 1;
    if (kt + 32 < DIMM)   // prefetch next tile into the other buffer
      stage_tiles(As[cur ^ 1], Bs[cur ^ 1], A, Bt, m0, n0, kt + 32, tid);

    v16bf afr[2], bfr[2];
    #pragma unroll
    for (int t = 0; t < 2; ++t) {
      int row = wm + t * 16 + nl;
      int col = wn + t * 16 + nl;
      #pragma unroll
      for (int v = 0; v < 8; ++v) {
        bf16x2 pa = *(const bf16x2*)&As[cur][row * 40 + a_kpat(v, hi)];
        afr[t][2 * v] = pa[0]; afr[t][2 * v + 1] = pa[1];
        bf16x2 pb = *(const bf16x2*)&Bs[cur][col * 40 + a_kpat(v, hi)];
        bfr[t][2 * v] = pb[0]; bfr[t][2 * v + 1] = pb[1];
      }
    }
    #pragma unroll
    for (int ti = 0; ti < 2; ++ti)
      #pragma unroll
      for (int tj = 0; tj < 2; ++tj)
        acc[ti][tj] = __builtin_amdgcn_wmma_f32_16x16x32_bf16(
            false, afr[ti], false, bfr[tj], (short)0, acc[ti][tj], false, false);

    ldsCopyWait();      // next tile landed
    __syncthreads();    // everyone done with cur + next staged
  }

  if (vmode == 0) {
    #pragma unroll
    for (int ti = 0; ti < 2; ++ti)
      #pragma unroll
      for (int tj = 0; tj < 2; ++tj)
        #pragma unroll
        for (int r = 0; r < 8; r += 2) {
          int gm = m0 + wm + ti * 16 + r + 8 * hi;
          int cn = n0 + wn + tj * 16 + nl;
          int h = cn >> 6, d = cn & 63;
          bf16x2 pk = f2bf2(acc[ti][tj][r] * scale, acc[ti][tj][r + 1] * scale);
          dst[((size_t)h * SEQ + gm) * DHEAD + d]     = pk[0];
          dst[((size_t)h * SEQ + gm + 1) * DHEAD + d] = pk[1];
        }
  } else {
    #pragma unroll
    for (int ti = 0; ti < 2; ++ti)
      #pragma unroll
      for (int tj = 0; tj < 2; ++tj)
        #pragma unroll
        for (int r = 0; r < 8; r += 2) {
          int gm = m0 + wm + ti * 16 + r + 8 * hi;
          int cn = n0 + wn + tj * 16 + nl;
          int h = cn >> 6, d = cn & 63;
          bf16x2 pk = f2bf2(acc[ti][tj][r], acc[ti][tj][r + 1]);
          dst[((size_t)h * DHEAD + d) * SEQ + gm]     = pk[0];
          dst[((size_t)h * DHEAD + d) * SEQ + gm + 1] = pk[1];
        }
  }
}

// ---------------------------------------------------------------------------
// Kernel 2: flash attention (causal).  grid = (SEQ/64, HEADS), 128 thr.
// Each wave owns 16 query rows; streams key blocks of 32.
// ---------------------------------------------------------------------------
__global__ __launch_bounds__(128) void flash_attn_kernel(
    const __bf16* __restrict__ qws, const __bf16* __restrict__ kws,
    const __bf16* __restrict__ vtws, __bf16* __restrict__ aows)
{
  __shared__ __bf16 Pl[4][16 * 32];   // per-wave P bounce (A-relayout)

  const int tid  = threadIdx.x;
  const int lane = tid & 31;
  const int w    = tid >> 5;
  const int nl   = lane & 15;
  const int hi   = lane >> 4;
  const int h    = blockIdx.y;
  const int q0   = blockIdx.x * 64 + w * 16;

  const __bf16* Qh  = qws  + (size_t)h * SEQ * DHEAD;
  const __bf16* Kh  = kws  + (size_t)h * SEQ * DHEAD;
  const __bf16* Vth = vtws + (size_t)h * DHEAD * SEQ;

  // Q A-fragments (two K-halves of d=64); beta pre-applied in Q
  v16bf qa[2];
  #pragma unroll
  for (int kk = 0; kk < 2; ++kk) {
    int row = q0 + nl;
    #pragma unroll
    for (int v = 0; v < 8; ++v) {
      int k = kk * 32 + a_kpat(v, hi);
      bf16x2 p = *(const bf16x2*)&Qh[(size_t)row * DHEAD + k];
      qa[kk][2 * v] = p[0]; qa[kk][2 * v + 1] = p[1];
    }
  }

  v8f o[4] = {};
  float mi[8], li[8];
  #pragma unroll
  for (int r = 0; r < 8; ++r) { mi[r] = -1e30f; li[r] = 0.0f; }

  for (int j0 = 0; j0 <= q0 + 15; j0 += 32) {
    // speculative prefetch of next K/V block (global_prefetch_b8; OOB-safe)
    __builtin_prefetch(&Kh[(size_t)(j0 + 32 + lane) * DHEAD], 0, 0);
    __builtin_prefetch(&Vth[(size_t)(lane * 2) * SEQ + j0 + 32], 0, 0);

    // ---- S = Q * K^T  (16 x 32 scores, two 16x16 tiles) ----
    v8f s[2];
    #pragma unroll
    for (int tj = 0; tj < 2; ++tj) {
      v8f st = {};
      int col = j0 + tj * 16 + nl;
      #pragma unroll
      for (int kk = 0; kk < 2; ++kk) {
        v16bf kb;
        #pragma unroll
        for (int v = 0; v < 8; ++v) {
          int k = kk * 32 + a_kpat(v, hi);
          bf16x2 p = *(const bf16x2*)&Kh[(size_t)col * DHEAD + k];
          kb[2 * v] = p[0]; kb[2 * v + 1] = p[1];
        }
        st = __builtin_amdgcn_wmma_f32_16x16x32_bf16(
            false, qa[kk], false, kb, (short)0, st, false, false);
      }
      s[tj] = st;
    }

    // ---- causal mask + online softmax ----
    #pragma unroll
    for (int r = 0; r < 8; ++r) {
      int mrow = q0 + r + 8 * hi;
      if (j0 + nl > mrow)      s[0][r] = -3.0e38f;
      if (j0 + 16 + nl > mrow) s[1][r] = -3.0e38f;
    }
    #pragma unroll
    for (int r = 0; r < 8; ++r) {
      float t = fmaxf(s[0][r], s[1][r]);
      #pragma unroll
      for (int off = 1; off < 16; off <<= 1)
        t = fmaxf(t, __shfl_xor(t, off, 32));     // 16-lane groups stay closed
      float mnew  = fmaxf(mi[r], t);
      float alpha = __expf(mi[r] - mnew);
      float p0 = __expf(s[0][r] - mnew);
      float p1 = __expf(s[1][r] - mnew);
      float rs = p0 + p1;
      #pragma unroll
      for (int off = 1; off < 16; off <<= 1)
        rs += __shfl_xor(rs, off, 32);
      li[r] = li[r] * alpha + rs;
      mi[r] = mnew;
      #pragma unroll
      for (int td = 0; td < 4; ++td) o[td][r] *= alpha;
      bf16x2 pk = f2bf2(p0, p1);
      Pl[w][(r + 8 * hi) * 32 + nl]      = pk[0];
      Pl[w][(r + 8 * hi) * 32 + 16 + nl] = pk[1];
    }

    // ---- P A-fragment from LDS (wave-private; DS ops in-order) ----
    v16bf pa;
    #pragma unroll
    for (int v = 0; v < 8; ++v) {
      bf16x2 p = *(const bf16x2*)&Pl[w][nl * 32 + a_kpat(v, hi)];
      pa[2 * v] = p[0]; pa[2 * v + 1] = p[1];
    }

    // ---- O += P * V  (V transposed: pairs contiguous along j) ----
    #pragma unroll
    for (int td = 0; td < 4; ++td) {
      v16bf vb;
      int col = td * 16 + nl;           // d index
      #pragma unroll
      for (int v = 0; v < 8; ++v) {
        int k = a_kpat(v, hi);   // j index
        bf16x2 p = *(const bf16x2*)&Vth[(size_t)col * SEQ + j0 + k];
        vb[2 * v] = p[0]; vb[2 * v + 1] = p[1];
      }
      o[td] = __builtin_amdgcn_wmma_f32_16x16x32_bf16(
          false, pa, false, vb, (short)0, o[td], false, false);
    }
  }

  // ---- normalize + write merged-head layout [n][h*64+d] ----
  #pragma unroll
  for (int td = 0; td < 4; ++td)
    #pragma unroll
    for (int r = 0; r < 8; r += 2) {
      int gm = q0 + r + 8 * hi;
      int cn = h * DHEAD + td * 16 + nl;
      bf16x2 pk = f2bf2(o[td][r] / li[r], o[td][r + 1] / li[r + 1]);
      aows[(size_t)gm * DIMM + cn]       = pk[0];
      aows[(size_t)(gm + 1) * DIMM + cn] = pk[1];
    }
}

// ---------------------------------------------------------------------------
// Kernel 3: output projection  out = attn @ Wo + bo   (f32 out)
// A = aows bf16 [4096][1024]; Wot = bf16 [n][k] (pre-transposed)
// ---------------------------------------------------------------------------
__global__ __launch_bounds__(128) void out_proj_kernel(
    const __bf16* __restrict__ A, const __bf16* __restrict__ Wot,
    const float* __restrict__ bo, float* __restrict__ out)
{
  __shared__ __bf16 As[2][64 * 40];
  __shared__ __bf16 Bs[2][64 * 40];

  const int tid  = threadIdx.x;
  const int lane = tid & 31;
  const int w    = tid >> 5;
  const int nl   = lane & 15;
  const int hi   = lane >> 4;
  const int m0   = blockIdx.y * 64;
  const int n0   = blockIdx.x * 64;
  const int wm   = (w >> 1) * 32;
  const int wn   = (w & 1) * 32;

  v8f acc[2][2] = {};

  stage_tiles(As[0], Bs[0], A, Wot, m0, n0, 0, tid);
  ldsCopyWait();
  __syncthreads();

  for (int kt = 0; kt < DIMM; kt += 32) {
    const int cur = (kt >> 5) & 1;
    if (kt + 32 < DIMM)
      stage_tiles(As[cur ^ 1], Bs[cur ^ 1], A, Wot, m0, n0, kt + 32, tid);

    v16bf afr[2], bfr[2];
    #pragma unroll
    for (int t = 0; t < 2; ++t) {
      int row = wm + t * 16 + nl;
      int col = wn + t * 16 + nl;
      #pragma unroll
      for (int v = 0; v < 8; ++v) {
        bf16x2 pa = *(const bf16x2*)&As[cur][row * 40 + a_kpat(v, hi)];
        afr[t][2 * v] = pa[0]; afr[t][2 * v + 1] = pa[1];
        bf16x2 pb = *(const bf16x2*)&Bs[cur][col * 40 + a_kpat(v, hi)];
        bfr[t][2 * v] = pb[0]; bfr[t][2 * v + 1] = pb[1];
      }
    }
    #pragma unroll
    for (int ti = 0; ti < 2; ++ti)
      #pragma unroll
      for (int tj = 0; tj < 2; ++tj)
        acc[ti][tj] = __builtin_amdgcn_wmma_f32_16x16x32_bf16(
            false, afr[ti], false, bfr[tj], (short)0, acc[ti][tj], false, false);

    ldsCopyWait();
    __syncthreads();
  }

  #pragma unroll
  for (int tj = 0; tj < 2; ++tj) {
    int cn = n0 + wn + tj * 16 + nl;
    float bias = bo[cn];
    #pragma unroll
    for (int ti = 0; ti < 2; ++ti)
      #pragma unroll
      for (int r = 0; r < 8; ++r) {
        int gm = m0 + wm + ti * 16 + r + 8 * hi;
        out[(size_t)gm * DIMM + cn] = acc[ti][tj][r] + bias;
      }
  }
}

// ---------------------------------------------------------------------------
extern "C" void kernel_launch(void* const* d_in, const int* in_sizes, int n_in,
                              void* d_out, int out_size, void* d_ws, size_t ws_size,
                              hipStream_t stream) {
  (void)in_sizes; (void)n_in; (void)out_size; (void)ws_size;
  const float* x  = (const float*)d_in[0];
  const float* Wq = (const float*)d_in[1];
  const float* Wk = (const float*)d_in[2];
  const float* Wv = (const float*)d_in[3];
  const float* Wo = (const float*)d_in[4];
  const float* bo = (const float*)d_in[5];
  float* out = (float*)d_out;

  const size_t nX = (size_t)SEQ * DIMM;           // 4M
  const size_t nW = (size_t)DIMM * DIMM;          // 1M
  const size_t nH = (size_t)HEADS * SEQ * DHEAD;  // 4M

  __bf16* xbf  = (__bf16*)d_ws;               // [4096][1024]
  __bf16* Wqt  = xbf  + nX;                   // [n][k]
  __bf16* Wkt  = Wqt  + nW;
  __bf16* Wvt  = Wkt  + nW;
  __bf16* Wot  = Wvt  + nW;
  __bf16* qws  = Wot  + nW;                   // [h][n][64]
  __bf16* kws  = qws  + nH;
  __bf16* vtws = kws  + nH;                   // [h][64][n]
  __bf16* aows = vtws + nH;                   // [4096][1024]

  // 0) one-time conversions
  cvt_bf16_kernel<<<dim3((int)(nX / 512)), 256, 0, stream>>>(x, xbf, (int)nX);
  cvt_transpose_kernel<<<dim3(32, 32), 256, 0, stream>>>(Wq, Wqt);
  cvt_transpose_kernel<<<dim3(32, 32), 256, 0, stream>>>(Wk, Wkt);
  cvt_transpose_kernel<<<dim3(32, 32), 256, 0, stream>>>(Wv, Wvt);
  cvt_transpose_kernel<<<dim3(32, 32), 256, 0, stream>>>(Wo, Wot);

  // 1) QKV projections (beta folded into Q)
  qkv_gemm_kernel<<<dim3(DIMM / 64, SEQ / 64), 128, 0, stream>>>(
      xbf, Wqt, qws, 0.125f, 0);
  qkv_gemm_kernel<<<dim3(DIMM / 64, SEQ / 64), 128, 0, stream>>>(
      xbf, Wkt, kws, 1.0f, 0);
  qkv_gemm_kernel<<<dim3(DIMM / 64, SEQ / 64), 128, 0, stream>>>(
      xbf, Wvt, vtws, 1.0f, 1);

  // 2) flash attention
  flash_attn_kernel<<<dim3(SEQ / 64, HEADS), 128, 0, stream>>>(
      qws, kws, vtws, aows);

  // 3) output projection
  out_proj_kernel<<<dim3(DIMM / 64, SEQ / 64), 128, 0, stream>>>(
      aows, Wot, bo, out);
}
